// TnlAttention_35424890257797
// MI455X (gfx1250) — matrix operand order repaired and checked by reading, not compiled, not run on hardware
//
#include <hip/hip_runtime.h>
#include <stdint.h>
#include <stddef.h>

typedef __bf16 bf16;
typedef __attribute__((ext_vector_type(16))) __bf16 v16bf;
typedef __attribute__((ext_vector_type(8)))  float  v8f;

#define WMMA_BF16(a, b, c) \
  __builtin_amdgcn_wmma_f32_16x16x32_bf16(false, (a), false, (b), (short)0, (c), false, false)

// Scheduler pipeline hints: group DS reads ahead of the WMMA chain so the
// hazard pass can emit partial s_wait_dscnt counts (loads retire in order).
//   mask 0x100 = DS read, mask 0x008 = MFMA/WMMA
#define SCHED_DS_THEN_WMMA(nds, nwmma)                       \
  do {                                                       \
    __builtin_amdgcn_sched_group_barrier(0x100, (nds), 0);   \
    __builtin_amdgcn_sched_group_barrier(0x008, (nwmma), 0); \
  } while (0)

// ---- problem constants (from reference setup_inputs) ----
static constexpr int BB = 2, NN = 4096, DD = 2048, HH = 16, HD = 128, CH = 128, NCH = 32;
static constexpr int MROWS = BB * NN;   // 8192

enum { MODE_SILU_BF16 = 0, MODE_F32 = 1, MODE_GATE_F32 = 2, MODE_BF16 = 3 };

union B16x8 { uint4 u; bf16 h[8]; };
union B16x4 { uint2 u; bf16 h[4]; };
union Frag  { uint4 u[2]; v16bf v; };

// ---- CDNA5 async global->LDS copy (ASYNCcnt-tracked), inline asm ----
__device__ __forceinline__ void async_load_b128(uint32_t lds_off, const void* gptr) {
  asm volatile("global_load_async_to_lds_b128 %0, %1, off"
               :: "v"(lds_off), "v"((uint64_t)(uintptr_t)gptr)
               : "memory");
}
__device__ __forceinline__ void wait_async0() {
  asm volatile("s_wait_asynccnt 0x0" ::: "memory");
}

// ============================================================================
// One-time conversion kernels
// ============================================================================
__global__ __launch_bounds__(256, 1)
void cvt_bf16_kernel(const float4* __restrict__ in, bf16* __restrict__ out, int nquads)
{
  int i = blockIdx.x * 256 + threadIdx.x;
  if (i >= nquads) return;
  float4 f = in[i];
  B16x4 t;
  t.h[0] = (bf16)f.x; t.h[1] = (bf16)f.y; t.h[2] = (bf16)f.z; t.h[3] = (bf16)f.w;
  *(uint2*)(out + (size_t)i * 4) = t.u;
}

// in[R][C] f32  ->  out[C][R] bf16   (tiled transpose)
__global__ __launch_bounds__(256, 1)
void transpose_cvt_kernel(const float* __restrict__ in, bf16* __restrict__ out, int R, int C)
{
  __shared__ float tile[32][33];
  int bx = blockIdx.x * 32;            // C base
  int by = blockIdx.y * 32;            // R base
  int tx = threadIdx.x & 31, ty = threadIdx.x >> 5;   // 32 x 8
  for (int i = 0; i < 32; i += 8)
    tile[ty + i][tx] = in[(size_t)(by + ty + i) * C + bx + tx];
  __syncthreads();
  for (int i = 0; i < 32; i += 8)
    out[(size_t)(bx + ty + i) * R + by + tx] = (bf16)tile[tx][ty + i];
}

// ============================================================================
// All-bf16 WMMA GEMM with double-buffered async LDS staging.
//   Out[m][n] = epilogue( sum_k A[m][k] * Wt[n][k] )    (Wt pre-transposed)
//   256 thr = 8 waves; 128x128 block tile; TILE_K = 64 (2 k-steps / stage).
// ============================================================================
__global__ __launch_bounds__(256, 1)
void wmma_gemm_kernel(const bf16* __restrict__ A, const bf16* __restrict__ Wt,
                      const float* __restrict__ Omul,
                      float* __restrict__ OutF, bf16* __restrict__ OutB,
                      int M, int K, int Nn, int mode)
{
  // smem[buf][0] = A tile [m][k] 128x64, smem[buf][1] = B tile [n][k] 128x64
  __shared__ __align__(16) bf16 smem[2][2][128][64];

  const int tid  = threadIdx.x;
  const int wv   = tid >> 5;
  const int lane = tid & 31;
  const int m0 = blockIdx.x * 128;
  const int n0 = blockIdx.y * 128;
  const int ka0   = (lane >> 4) << 3;   // K-octet (0 or 8), unused: A frag uses khalf
  const int khalf = (lane >> 4) << 4;   // B frag: K-half select  (0 or 16)
  const int lrow  = (lane >> 4) << 3;   // C/D: row block (0 or 8)
  const int lcol  = lane & 15;

  const uint32_t ldsA0 = (uint32_t)(uintptr_t)&smem[0][0][0][0];
  const uint32_t ldsB0 = (uint32_t)(uintptr_t)&smem[0][1][0][0];
  const uint32_t tileBytes = 128u * 64u * sizeof(bf16);    // 16 KB
  const uint32_t bufStride = 2u * tileBytes;               // per [buf] step

  v8f acc[8];
  { v8f z{}; for (int j = 0; j < 8; ++j) acc[j] = z; }

  const int nst = K >> 6;

  // issue async copies for stage s into buffer buf (8 instrs / wave)
  auto issue = [&](int s, int buf) {
    const int k0 = s << 6;
    const char* Ab = (const char*)A  + ((size_t)m0 * K + k0) * sizeof(bf16);
    const char* Bb = (const char*)Wt + ((size_t)n0 * K + k0) * sizeof(bf16);
    const uint32_t bufOff = (uint32_t)buf * bufStride;
#pragma unroll
    for (int t = 0; t < 4; ++t) {
      uint32_t off = (uint32_t)((wv * 4 + t) * 512 + lane * 16); // byte in tile
      uint32_t r = off >> 7, cb = off & 127;                     // row, col byte
      size_t g = (size_t)r * K * sizeof(bf16) + cb;
      async_load_b128(ldsA0 + bufOff + off, Ab + g);
      async_load_b128(ldsB0 + bufOff + off, Bb + g);
    }
  };

  issue(0, 0);
  wait_async0();
  __syncthreads();

  for (int s = 0; s < nst; ++s) {
    const int buf = s & 1;
    if (s + 1 < nst) issue(s + 1, buf ^ 1);

#pragma unroll
    for (int kk = 0; kk < 2; ++kk) {
      // ---- A fragment (16x32) ----
      Frag a;
      {
        const bf16* p = &smem[buf][0][wv * 16 + lcol][kk * 32];
        a.u[0] = *(const uint4*)(p + khalf);
        a.u[1] = *(const uint4*)(p + khalf + 8);
      }
      // ---- preload all 8 B fragments, then WMMA back-to-back ----
      Frag bfr[8];
#pragma unroll
      for (int j = 0; j < 8; ++j) {
        const bf16* p = &smem[buf][1][j * 16 + lcol][kk * 32 + khalf];
        bfr[j].u[0] = *(const uint4*)p;
        bfr[j].u[1] = *(const uint4*)(p + 8);
      }
#pragma unroll
      for (int j = 0; j < 8; ++j)
        acc[j] = WMMA_BF16(a.v, bfr[j].v, acc[j]);

      // pipeline: 18 DS reads (2 A + 16 B) ahead of the 8 WMMAs
      SCHED_DS_THEN_WMMA(18, 8);
    }

    if (s + 1 < nst) {
      wait_async0();      // own stage-(s+1) copies done (overlapped with WMMA above)
      __syncthreads();    // everyone finished compute(s) and wait
    }
  }

  // ---- epilogue ----
  for (int j = 0; j < 8; ++j) {
    for (int vv = 0; vv < 8; ++vv) {
      int r = m0 + wv * 16 + lrow + vv;
      int c = n0 + j * 16 + lcol;
      size_t idx = (size_t)r * Nn + c;
      float val = acc[j][vv];
      if (mode == MODE_SILU_BF16) {
        OutB[idx] = (bf16)(val / (1.f + __expf(-val)));
      } else if (mode == MODE_BF16) {
        OutB[idx] = (bf16)val;
      } else if (mode == MODE_GATE_F32) {
        float g = 1.f / (1.f + __expf(-val));
        OutF[idx] = g * Omul[idx];
      } else {
        OutF[idx] = val;
      }
    }
  }
}

// ============================================================================
// Lightning attention: one workgroup per (b, h); sequential scan over 32
// chunks of 128 tokens. Per-wave register state S; k staged via async copies,
// v staged transposed; sb reused: scores[c][s] then S^T[e][d] snapshot.
// ============================================================================
__global__ __launch_bounds__(256, 1)
void lightning_attn_kernel(const bf16* __restrict__ qg, const bf16* __restrict__ kg,
                           const bf16* __restrict__ vg, const float* __restrict__ ld,
                           float* __restrict__ og)
{
  __shared__ __align__(16) bf16 ks[128][128];    // [s][d]   32 KB
  __shared__ __align__(16) bf16 vst[128][128];   // [e][s]   32 KB (v transposed)
  __shared__ __align__(16) bf16 sb[128][128];    // 32 KB, dual-purpose

  const int tid  = threadIdx.x;
  const int wv   = tid >> 5;
  const int lane = tid & 31;
  const int h = blockIdx.x & (HH - 1);
  const int b = blockIdx.x >> 4;
  const float ldh  = ld[h];
  const float cdec = __expf(ldh * (float)CH);
  const int ka0   = (lane >> 4) << 3;
  const int khalf = (lane >> 4) << 4;
  const int lrow  = (lane >> 4) << 3;
  const int lcol  = lane & 15;
  const uint32_t ksBase = (uint32_t)(uintptr_t)&ks[0][0];

  v8f S[8];
  { v8f z{}; for (int j = 0; j < 8; ++j) S[j] = z; }

  for (int ch = 0; ch < NCH; ++ch) {
    const size_t tbase = (size_t)(b * NN + ch * CH) * DD + (size_t)h * HD;

    // ---- async-stage k rows (128 x 256B), 8 instrs / wave ----
#pragma unroll
    for (int t = 0; t < 8; ++t) {
      uint32_t off = (uint32_t)(((wv * 8 + t) << 9) + (lane << 4));
      uint32_t r = off >> 8, cb = off & 255;
      async_load_b128(ksBase + off,
                      (const char*)(kg + tbase) + (size_t)r * DD * sizeof(bf16) + cb);
    }
    // ---- stage v transposed (manual scatter) ----
    for (int i = tid; i < 2048; i += 256) {
      int r  = i >> 4;
      int c8 = (i & 15) << 3;
      B16x8 t; t.u = *(const uint4*)(vg + tbase + (size_t)r * DD + c8);
      for (int e2 = 0; e2 < 8; ++e2) vst[c8 + e2][r] = t.h[e2];
    }
    // ---- q fragments in registers (wave rows c = wv*16 + lcol) ----
    v16bf qf[4];
    {
      const bf16* qrow = qg + tbase + (size_t)(wv * 16 + lcol) * DD;
      for (int kk = 0; kk < 4; ++kk) {
        Frag t;
        t.u[0] = *(const uint4*)(qrow + kk * 32 + khalf);
        t.u[1] = *(const uint4*)(qrow + kk * 32 + khalf + 8);
        qf[kk] = t.v;
      }
    }
    wait_async0();
    __syncthreads();

    // ---- phase 1: scores = q @ k^T (K-dim = feature d), decay, -> sb ----
    {
      v8f sc[8]; { v8f z{}; for (int j = 0; j < 8; ++j) sc[j] = z; }
      for (int kk = 0; kk < 4; ++kk) {
        for (int j = 0; j < 8; ++j) {
          Frag bfr;                       // B[d][s] = k[s][d] -> row read of ks
          const bf16* p = &ks[j * 16 + lcol][kk * 32 + khalf];
          bfr.u[0] = *(const uint4*)p; bfr.u[1] = *(const uint4*)(p + 8);
          sc[j] = WMMA_BF16(qf[kk], bfr.v, sc[j]);
        }
        SCHED_DS_THEN_WMMA(16, 8);
      }
      for (int j = 0; j < 8; ++j)
        for (int vv = 0; vv < 8; ++vv) {
          int c = wv * 16 + lrow + vv;
          int s = j * 16 + lcol;
          float f = (c >= s) ? __expf(ldh * (float)(c - s)) : 0.f;
          sb[c][s] = (bf16)(sc[j][vv] * f);
        }
    }
    __syncthreads();

    // ---- phase 2: o_intra = scores @ v (K-dim = token s) ----
    v8f oacc[8]; { v8f z{}; for (int j = 0; j < 8; ++j) oacc[j] = z; }
    for (int kk = 0; kk < 4; ++kk) {
      Frag a;
      const bf16* ap = &sb[wv * 16 + lcol][kk * 32];
      a.u[0] = *(const uint4*)(ap + khalf);
      a.u[1] = *(const uint4*)(ap + khalf + 8);
      for (int j = 0; j < 8; ++j) {
        Frag bfr;                          // B[s][e] = vst[e][s]
        const bf16* p = &vst[j * 16 + lcol][kk * 32 + khalf];
        bfr.u[0] = *(const uint4*)p; bfr.u[1] = *(const uint4*)(p + 8);
        oacc[j] = WMMA_BF16(a.v, bfr.v, oacc[j]);
      }
      SCHED_DS_THEN_WMMA(18, 8);
    }
    __syncthreads();

    // ---- snapshot S^T (bf16) into sb for cross-wave q@S ----
    for (int j = 0; j < 8; ++j)
      for (int vv = 0; vv < 8; ++vv) {
        int d = wv * 16 + lrow + vv;
        int e = j * 16 + lcol;
        sb[e][d] = (bf16)S[j][vv];
      }
    __syncthreads();

    // ---- phase 3: o_inter = (q @ S) * q_decay(c), folded into oacc ----
    {
      v8f ai[8]; { v8f z{}; for (int j = 0; j < 8; ++j) ai[j] = z; }
      for (int kk = 0; kk < 4; ++kk) {
        for (int j = 0; j < 8; ++j) {
          Frag bfr;                        // B[d][e] = sb[e][d]
          const bf16* p = &sb[j * 16 + lcol][kk * 32 + khalf];
          bfr.u[0] = *(const uint4*)p; bfr.u[1] = *(const uint4*)(p + 8);
          ai[j] = WMMA_BF16(qf[kk], bfr.v, ai[j]);
        }
        SCHED_DS_THEN_WMMA(16, 8);
      }
      for (int vv = 0; vv < 8; ++vv) {
        int c = wv * 16 + lrow + vv;
        float qd = __expf(ldh * (float)(c + 1));
        for (int j = 0; j < 8; ++j) oacc[j][vv] += qd * ai[j][vv];
      }
    }

    // ---- store output chunk (f32) ----
    for (int j = 0; j < 8; ++j)
      for (int vv = 0; vv < 8; ++vv) {
        int c = wv * 16 + lrow + vv;
        int e = j * 16 + lcol;
        og[tbase + (size_t)c * DD + e] = oacc[j][vv];
      }

    // ---- phase 4: S = cdec*S + (k * k_decay)^T @ v ----
    for (int j = 0; j < 8; ++j)
      for (int vv = 0; vv < 8; ++vv) S[j][vv] *= cdec;
    for (int kk = 0; kk < 4; ++kk) {
      Frag a;                              // A[d][s] = ks[s][d] * kdec(s)
      int d = wv * 16 + lcol;
      for (int e2 = 0; e2 < 16; ++e2) {
        int s = kk * 32 + khalf + e2;
        float kd = __expf(ldh * (float)(CH - 1 - s));
        a.v[e2] = (bf16)((float)ks[s][d] * kd);
      }
      for (int j = 0; j < 8; ++j) {
        Frag bfr;
        const bf16* p = &vst[j * 16 + lcol][kk * 32 + khalf];
        bfr.u[0] = *(const uint4*)p; bfr.u[1] = *(const uint4*)(p + 8);
        S[j] = WMMA_BF16(a.v, bfr.v, S[j]);
      }
    }
    __syncthreads();   // protect ks/vst before next chunk's staging
  }
}

// ============================================================================
// Per-(row, head) group RMSNorm: one wave per group, shuffle reduction.
// ============================================================================
__global__ __launch_bounds__(256, 1)
void rmsnorm_kernel(const float* __restrict__ og, const float* __restrict__ wnorm,
                    bf16* __restrict__ on)
{
  int gw   = blockIdx.x * 8 + (threadIdx.x >> 5);
  int lane = threadIdx.x & 31;
  int row = gw >> 4;
  int h   = gw & 15;
  const float* p = og + (size_t)row * DD + (size_t)h * HD;

  float x[4];
  float ss = 0.f;
  for (int i = 0; i < 4; ++i) { x[i] = p[lane + i * 32]; ss += x[i] * x[i]; }
  for (int off = 16; off > 0; off >>= 1) ss += __shfl_xor(ss, off, 32);
  float r = rsqrtf(ss * (1.0f / (float)HD) + 1e-5f);

  for (int i = 0; i < 4; ++i) {
    int c = lane + i * 32;
    on[(size_t)row * DD + h * HD + c] = (bf16)(x[i] * r * wnorm[h * HD + c]);
  }
}

// ============================================================================
// Orchestration
// ============================================================================
extern "C" void kernel_launch(void* const* d_in, const int* in_sizes, int n_in,
                              void* d_out, int out_size, void* d_ws, size_t ws_size,
                              hipStream_t stream)
{
  (void)in_sizes; (void)n_in; (void)out_size; (void)ws_size;
  const float* x   = (const float*)d_in[0];
  const float* ldk = (const float*)d_in[1];
  const float* Wq  = (const float*)d_in[2];
  const float* Wk  = (const float*)d_in[3];
  const float* Wv  = (const float*)d_in[4];
  const float* Wo  = (const float*)d_in[5];
  const float* Wg1 = (const float*)d_in[6];
  const float* Wg2 = (const float*)d_in[7];
  const float* nw  = (const float*)d_in[8];
  float* out = (float*)d_out;

  // ---- workspace carve ----
  char* p = (char*)d_ws;
  bf16*  xb  = (bf16*)p;  p += (size_t)MROWS * DD * sizeof(bf16);
  bf16*  qb  = (bf16*)p;  p += (size_t)MROWS * DD * sizeof(bf16);
  bf16*  kb  = (bf16*)p;  p += (size_t)MROWS * DD * sizeof(bf16);
  bf16*  vb  = (bf16*)p;  p += (size_t)MROWS * DD * sizeof(bf16);
  bf16*  g1b = (bf16*)p;  p += (size_t)MROWS * HD * sizeof(bf16);
  float* o   = (float*)p; p += (size_t)MROWS * DD * sizeof(float);
  bf16*  on  = (bf16*)p;  p += (size_t)MROWS * DD * sizeof(bf16);
  bf16*  Wqt = (bf16*)p;  p += (size_t)DD * DD * sizeof(bf16);
  bf16*  Wkt = (bf16*)p;  p += (size_t)DD * DD * sizeof(bf16);
  bf16*  Wvt = (bf16*)p;  p += (size_t)DD * DD * sizeof(bf16);
  bf16*  Wot = (bf16*)p;  p += (size_t)DD * DD * sizeof(bf16);
  bf16*  Wg1t = (bf16*)p; p += (size_t)HD * DD * sizeof(bf16);   // [128][2048]
  bf16*  Wg2t = (bf16*)p;                                        // [2048][128]

  dim3 blk(256);
  dim3 gFull(MROWS / 128, DD / 128);   // 64 x 16
  dim3 gG1(MROWS / 128, HD / 128);     // 64 x 1
  dim3 gT(DD / 32, DD / 32);           // 64 x 64

  // ---- one-time bf16 conversions / weight transposes ----
  cvt_bf16_kernel<<<MROWS * DD / 4 / 256, blk, 0, stream>>>((const float4*)x, xb,
                                                            MROWS * DD / 4);
  transpose_cvt_kernel<<<gT, blk, 0, stream>>>(Wq, Wqt, DD, DD);
  transpose_cvt_kernel<<<gT, blk, 0, stream>>>(Wk, Wkt, DD, DD);
  transpose_cvt_kernel<<<gT, blk, 0, stream>>>(Wv, Wvt, DD, DD);
  transpose_cvt_kernel<<<gT, blk, 0, stream>>>(Wo, Wot, DD, DD);
  transpose_cvt_kernel<<<dim3(HD / 32, DD / 32), blk, 0, stream>>>(Wg1, Wg1t, DD, HD);
  transpose_cvt_kernel<<<dim3(DD / 32, HD / 32), blk, 0, stream>>>(Wg2, Wg2t, HD, DD);

  // ---- q/k/v = silu(x @ W*) -> bf16 ----
  wmma_gemm_kernel<<<gFull, blk, 0, stream>>>(xb, Wqt, nullptr, nullptr, qb,
                                              MROWS, DD, DD, MODE_SILU_BF16);
  wmma_gemm_kernel<<<gFull, blk, 0, stream>>>(xb, Wkt, nullptr, nullptr, kb,
                                              MROWS, DD, DD, MODE_SILU_BF16);
  wmma_gemm_kernel<<<gFull, blk, 0, stream>>>(xb, Wvt, nullptr, nullptr, vb,
                                              MROWS, DD, DD, MODE_SILU_BF16);
  // ---- g1 = x @ Wg1 -> bf16 ----
  wmma_gemm_kernel<<<gG1, blk, 0, stream>>>(xb, Wg1t, nullptr, nullptr, g1b,
                                            MROWS, DD, HD, MODE_BF16);
  // ---- o = lightning_attn(q, k, v, ld) -> f32 ----
  lightning_attn_kernel<<<BB * HH, blk, 0, stream>>>(qb, kb, vb, ldk, o);
  // ---- o = o * sigmoid(g1 @ Wg2)  (in place) ----
  wmma_gemm_kernel<<<gFull, blk, 0, stream>>>(g1b, Wg2t, o, o, nullptr,
                                              MROWS, HD, DD, MODE_GATE_F32);
  // ---- on = group_rmsnorm(o) * w -> bf16 ----
  rmsnorm_kernel<<<MROWS * HH / 8, blk, 0, stream>>>(o, nw, on);
  // ---- out = on @ Wo -> f32 ----
  wmma_gemm_kernel<<<gFull, blk, 0, stream>>>(on, Wot, nullptr, out, nullptr,
                                              MROWS, DD, DD, MODE_F32);
}
